// NetMamba_81535659148007
// MI455X (gfx1250) — matrix operand order repaired and checked by reading, not compiled, not run on hardware
//
#include <hip/hip_runtime.h>
#include <hip/hip_bf16.h>
#include <math.h>

// ---------------- model dims ----------------
#define BATCH    32
#define LSEQ     401
#define ML       (BATCH * LSEQ)      // 12832 rows (= 802 * 16, WMMA-friendly)
#define DMODEL   192
#define DINNER   384
#define DSTATE   16
#define DTRANK   12
#define XPROJ_N  (DTRANK + 2 * DSTATE)   // 44
#define DTK      32                       // dt_proj K padded 12 -> 32
#define EPSV     1e-5f

typedef __attribute__((ext_vector_type(16))) _Float16 v16h;
typedef __attribute__((ext_vector_type(8)))  _Float16 v8h;
typedef __attribute__((ext_vector_type(8)))  float    v8f;

// =====================================================================
// WMMA GEMM (all-f16 operands):  D[M,N] = A[M,K] * W[N,K]^T (+ bias[N])
// Row-major f16 A (lda) and W (ldw), f32 D (ldd).
// REQUIRES: K % 32 == 0, M % 16 == 0, lda/ldw % 8 == 0 (16B alignment).
// One wave32 computes a 16x64 strip (4 N-subtiles): the A fragment is
// reused across 4 v_wmma_f32_16x16x32_f16 per k-step; the inner loop is
// pure b128 loads + register concat + WMMA (no converts, no guards).
// OOB N columns compute on clamped-row garbage but are never stored.
// =====================================================================
__global__ void wmma_gemm(const _Float16* __restrict__ A,
                          const _Float16* __restrict__ W,
                          const float* __restrict__ bias,
                          float* __restrict__ D,
                          int M, int N, int K,
                          int lda, int ldw, int ldd)
{
    const int lane    = threadIdx.x & 31;
    const int wave    = threadIdx.x >> 5;
    const int tilesN4 = (N + 63) >> 6;         // groups of four 16-col tiles
    const int tilesM  = (M + 15) >> 4;
    const int wtile   = blockIdx.x * (blockDim.x >> 5) + wave;
    if (wtile >= tilesM * tilesN4) return;     // wave-uniform: EXEC stays full

    const int tm = wtile / tilesN4;
    const int tg = wtile % tilesN4;
    const int m0 = tm << 4;
    const int n0 = tg << 6;

    const int hs   = lane >> 4;      // lane-group select (0/1)
    const int rsel = lane & 15;

    const _Float16* __restrict__ arow = A + (long)(m0 + rsel) * lda;

    int ncol[4];
    const _Float16* wrow[4];
#pragma unroll
    for (int t = 0; t < 4; ++t) {
        ncol[t] = n0 + t * 16 + rsel;
        const int safe = (ncol[t] < N) ? ncol[t] : (N - 1);   // stay in-bounds
        wrow[t] = W + (long)safe * ldw;
    }

    v8f c[4] = {{}, {}, {}, {}};

    for (int k0 = 0; k0 < K; k0 += 32) {
        // A 16x32 f16 frag: halfs 0..7 -> K = k0+8*hs+j ; 8..15 -> +16
        const _Float16* ap = arow + k0 + 8 * hs;
        const v8h alo = *(const v8h*)(ap);
        const v8h ahi = *(const v8h*)(ap + 16);
        const v16h a = __builtin_shufflevector(
            alo, ahi, 0,1,2,3,4,5,6,7,8,9,10,11,12,13,14,15);

#pragma unroll
        for (int t = 0; t < 4; ++t) {
            // B 32x16 f16 frag: lane owns col ncol[t], halfs 0..7 -> K=k0+8*hs+j ; 8..15 -> +16
            const _Float16* wp = wrow[t] + k0 + 8 * hs;
            const v8h blo = *(const v8h*)(wp);
            const v8h bhi = *(const v8h*)(wp + 16);
            const v16h b = __builtin_shufflevector(
                blo, bhi, 0,1,2,3,4,5,6,7,8,9,10,11,12,13,14,15);

            c[t] = __builtin_amdgcn_wmma_f32_16x16x32_f16(
                       false, a, false, b, (short)0, c[t], false, false);
        }
    }

    // store: D layout lane=col, VGPR v -> M = m0 + 8*hs + v (always < M)
#pragma unroll
    for (int t = 0; t < 4; ++t) {
        if (ncol[t] < N) {
            const float bv = bias ? bias[ncol[t]] : 0.0f;
#pragma unroll
            for (int v = 0; v < 8; ++v) {
                const int m = m0 + 8 * hs + v;
                D[(long)m * ldd + ncol[t]] = c[t][v] + bv;
            }
        }
    }
}

// =====================================================================
// f32 -> f16 weight conversion (contiguous)
// =====================================================================
__global__ void cvt_f16_kernel(const float* __restrict__ src,
                               _Float16* __restrict__ dst, int n)
{
    const int i = blockIdx.x * blockDim.x + threadIdx.x;
    if (i < n) dst[i] = (_Float16)src[i];
}

// =====================================================================
// f32 -> f16 with row re-stride + zero K padding (dt_w 12->32, xdbl 44->32)
// =====================================================================
__global__ void pad_cvt_kernel(const float* __restrict__ src,
                               _Float16* __restrict__ dst,
                               int rows, int ldsrc, int kin, int kout)
{
    const int i = blockIdx.x * blockDim.x + threadIdx.x;
    if (i >= rows * kout) return;
    const int k = i % kout;
    const int r = i / kout;
    dst[i] = (k < kin) ? (_Float16)src[(long)r * ldsrc + k] : (_Float16)0.0f;
}

// =====================================================================
// Patch embed + positional embed + cls token -> x[B, L, DMODEL] (f32)
// =====================================================================
__global__ void patch_embed_kernel(const float* __restrict__ imgs,
                                   const float* __restrict__ pw,
                                   const float* __restrict__ pb,
                                   const float* __restrict__ pos,
                                   const float* __restrict__ cls,
                                   float* __restrict__ x)
{
    const int idx = blockIdx.x * blockDim.x + threadIdx.x;
    if (idx >= ML * DMODEL) return;
    const int d = idx % DMODEL;
    const int m = idx / DMODEL;
    const int l = m % LSEQ;
    const int b = m / LSEQ;
    float v;
    if (l < LSEQ - 1) {
        const float* p = imgs + (long)b * 1600 + (long)l * 4;
        v = pb[d] + pos[(long)l * DMODEL + d];
#pragma unroll
        for (int s = 0; s < 4; ++s) v += p[s] * pw[s * DMODEL + d];
    } else {
        v = cls[d] + pos[(long)(LSEQ - 1) * DMODEL + d];
    }
    x[(long)m * DMODEL + d] = v;
}

// =====================================================================
// residual = hidden (+ resin) [f32];  h = rmsnorm(residual)*w -> f16
// =====================================================================
__global__ void add_rmsnorm_kernel(const float* __restrict__ hid,
                                   const float* __restrict__ resin,
                                   float* __restrict__ resout,
                                   _Float16* __restrict__ hout,
                                   const float* __restrict__ w)
{
    __shared__ float sm[256];
    const int row = blockIdx.x;
    const int t   = threadIdx.x;
    float v = 0.0f;
    if (t < DMODEL) {
        v = hid[(long)row * DMODEL + t];
        if (resin) v += resin[(long)row * DMODEL + t];
        resout[(long)row * DMODEL + t] = v;
    }
    sm[t] = v * v;
    __syncthreads();
    for (int s = 128; s > 0; s >>= 1) {
        if (t < s) sm[t] += sm[t + s];
        __syncthreads();
    }
    const float inv = rsqrtf(sm[0] * (1.0f / DMODEL) + EPSV);
    if (t < DMODEL) hout[(long)row * DMODEL + t] = (_Float16)(v * inv * w[t]);
}

// =====================================================================
// Depthwise causal conv (width 4) + SiLU.  u from xz[:, 0:384] (ld 768).
// Writes f32 (for the scan) and f16 (for the x_proj GEMM).
// =====================================================================
__global__ void conv_silu_kernel(const float* __restrict__ xz,
                                 const float* __restrict__ cw,
                                 const float* __restrict__ cb,
                                 float* __restrict__ u2,
                                 _Float16* __restrict__ u16)
{
    const int idx = blockIdx.x * blockDim.x + threadIdx.x;
    if (idx >= ML * DINNER) return;
    const int d = idx % DINNER;
    const int m = idx / DINNER;
    const int l = m % LSEQ;
    float acc = cb[d];
#pragma unroll
    for (int j = 0; j < 4; ++j) {
        const int ls = l - 3 + j;
        if (ls >= 0)
            acc += cw[d * 4 + j] * xz[(long)(m - l + ls) * (2 * DINNER) + d];
    }
    const float s = acc / (1.0f + __expf(-acc));   // silu
    u2[(long)m * DINNER + d]  = s;
    u16[(long)m * DINNER + d] = (_Float16)s;
}

// =====================================================================
// Selective scan: one thread per (batch, channel); 16-state recurrence.
// Reads f32 u/B/C/dt, writes gated y as f16 for the out_proj GEMM.
// =====================================================================
__global__ void mamba_scan_kernel(const float* __restrict__ dtf,
                                  const float* __restrict__ dtb,
                                  const float* __restrict__ xz,   // z at col 384
                                  const float* __restrict__ u2,
                                  const float* __restrict__ xdbl, // B at 12, C at 28
                                  const float* __restrict__ A_log,
                                  const float* __restrict__ Dv,
                                  _Float16* __restrict__ y16)
{
    const int idx = blockIdx.x * blockDim.x + threadIdx.x;
    if (idx >= BATCH * DINNER) return;
    const int d = idx % DINNER;
    const int b = idx / DINNER;

    float Am[DSTATE], h[DSTATE];
#pragma unroll
    for (int n = 0; n < DSTATE; ++n) {
        Am[n] = -__expf(A_log[(long)d * DSTATE + n]);
        h[n]  = 0.0f;
    }
    const float dbias = dtb[d];
    const float dval  = Dv[d];

    for (int l = 0; l < LSEQ; ++l) {
        const long  m  = (long)b * LSEQ + l;
        const float xr = dtf[m * DINNER + d] + dbias;
        const float dt = (xr > 20.0f) ? xr : log1pf(__expf(xr)); // softplus
        const float u  = u2[m * DINNER + d];
        const float* bc = xdbl + m * XPROJ_N;
        float y = 0.0f;
#pragma unroll
        for (int n = 0; n < DSTATE; ++n) {
            const float dA = __expf(dt * Am[n]);
            h[n] = dA * h[n] + dt * bc[DTRANK + n] * u;
            y += h[n] * bc[DTRANK + DSTATE + n];
        }
        y += u * dval;
        const float z = xz[m * (2 * DINNER) + DINNER + d];
        y *= z / (1.0f + __expf(-z));    // * silu(z)
        y16[m * DINNER + d] = (_Float16)y;
    }
}

// =====================================================================
// Gather cls rows (l = L-1) into a compact 32 x 192 f16 matrix
// =====================================================================
__global__ void gather_cls_kernel(const _Float16* __restrict__ x,
                                  _Float16* __restrict__ clsbuf)
{
    const int idx = blockIdx.x * blockDim.x + threadIdx.x;
    if (idx >= BATCH * DMODEL) return;
    const int d = idx % DMODEL;
    const int b = idx / DMODEL;
    clsbuf[(long)b * DMODEL + d] =
        x[((long)b * LSEQ + (LSEQ - 1)) * DMODEL + d];
}

// =====================================================================
// Host launcher
// =====================================================================
static inline int gemm_grid(int M, int N)
{
    const int tiles = ((M + 15) / 16) * ((N + 63) / 64);
    return (tiles + 7) / 8;   // 8 waves per 256-thread block
}

extern "C" void kernel_launch(void* const* d_in, const int* in_sizes, int n_in,
                              void* d_out, int out_size, void* d_ws, size_t ws_size,
                              hipStream_t stream)
{
    const float* imgs      = (const float*)d_in[0];
    const float* patch_w   = (const float*)d_in[1];
    const float* patch_b   = (const float*)d_in[2];
    const float* pos_embed = (const float*)d_in[3];
    const float* cls_token = (const float*)d_in[4];
    const float* norm_w    = (const float*)d_in[5];
    const float* in_proj_w = (const float*)d_in[6];
    const float* conv_w    = (const float*)d_in[7];
    const float* conv_b    = (const float*)d_in[8];
    const float* xproj_w   = (const float*)d_in[9];
    const float* dt_w      = (const float*)d_in[10];
    const float* dt_b      = (const float*)d_in[11];
    const float* A_log     = (const float*)d_in[12];
    const float* D_param   = (const float*)d_in[13];
    const float* out_w     = (const float*)d_in[14];
    const float* normf_w   = (const float*)d_in[15];
    const float* head_w    = (const float*)d_in[16];
    const float* head_b    = (const float*)d_in[17];
    float* out = (float*)d_out;

    // ---- workspace layout: f32 regions first, then 16B-aligned f16 ----
    float* f   = (float*)d_ws;
    float* hid   = f;                           // ML*192
    float* resid = hid   + (long)ML * DMODEL;   // ML*192
    float* xz    = resid + (long)ML * DMODEL;   // ML*768
    float* u2    = xz    + (long)ML * 2 * DINNER; // ML*384
    float* dtf   = u2    + (long)ML * DINNER;   // ML*384
    float* xdbl  = dtf   + (long)ML * DINNER;   // ML*44 used (48 reserved)

    _Float16* hbuf16 = (_Float16*)(xdbl + (long)ML * 48);
    _Float16* u16    = hbuf16 + (long)ML * DMODEL;   // ML*384
    _Float16* y16    = u16    + (long)ML * DINNER;   // ML*384
    _Float16* dtp16  = y16    + (long)ML * DINNER;   // ML*32 (padded dt A)
    _Float16* cls16  = dtp16  + (long)ML * DTK;      // 32*192
    _Float16* inw16  = cls16  + (long)BATCH * DMODEL;
    _Float16* xpw16  = inw16  + (long)4 * 2 * DINNER * DMODEL;  // 589824
    _Float16* dtw16  = xpw16  + (long)4 * XPROJ_N * DINNER;     // 67584
    _Float16* outw16 = dtw16  + (long)4 * DINNER * DTK;         // 49152
    _Float16* hdw16  = outw16 + (long)4 * DMODEL * DINNER;      // 294912

    const int ew = 256;
    const int nInW = 4 * 2 * DINNER * DMODEL;
    const int nXpW = 4 * XPROJ_N * DINNER;
    const int nOW  = 4 * DMODEL * DINNER;
    const int nHW  = 1000 * DMODEL;

    // 0) one-time (per launch) weight conversions to f16
    cvt_f16_kernel<<<(nInW + ew - 1) / ew, ew, 0, stream>>>(in_proj_w, inw16, nInW);
    cvt_f16_kernel<<<(nXpW + ew - 1) / ew, ew, 0, stream>>>(xproj_w, xpw16, nXpW);
    cvt_f16_kernel<<<(nOW  + ew - 1) / ew, ew, 0, stream>>>(out_w, outw16, nOW);
    cvt_f16_kernel<<<(nHW  + ew - 1) / ew, ew, 0, stream>>>(head_w, hdw16, nHW);
    pad_cvt_kernel<<<(4 * DINNER * DTK + ew - 1) / ew, ew, 0, stream>>>(
        dt_w, dtw16, 4 * DINNER, DTRANK, DTRANK, DTK);   // 12 -> 32 zero-pad

    // 1) patch + pos + cls embed -> hid (f32)
    patch_embed_kernel<<<(ML * DMODEL + ew - 1) / ew, ew, 0, stream>>>(
        imgs, patch_w, patch_b, pos_embed, cls_token, hid);

    for (int i = 0; i < 4; ++i) {
        // 2) residual add + RMSNorm -> hbuf16 (f16)
        add_rmsnorm_kernel<<<ML, 256, 0, stream>>>(
            hid, (i == 0) ? nullptr : resid, resid, hbuf16,
            norm_w + (long)i * DMODEL);

        // 3) in_proj: xz[ML,768] = hbuf16 @ in_w^T   (WMMA, K=192)
        wmma_gemm<<<gemm_grid(ML, 2 * DINNER), 256, 0, stream>>>(
            hbuf16, inw16 + (long)i * 2 * DINNER * DMODEL, nullptr, xz,
            ML, 2 * DINNER, DMODEL, DMODEL, DMODEL, 2 * DINNER);

        // 4) depthwise conv + silu -> u2 (f32) + u16 (f16)
        conv_silu_kernel<<<(ML * DINNER + ew - 1) / ew, ew, 0, stream>>>(
            xz, conv_w + (long)i * DINNER * 4, conv_b + (long)i * DINNER,
            u2, u16);

        // 5) x_proj: xdbl[ML,44] = u16 @ xp_w^T   (WMMA, K=384, N padded)
        wmma_gemm<<<gemm_grid(ML, XPROJ_N), 256, 0, stream>>>(
            u16, xpw16 + (long)i * XPROJ_N * DINNER, nullptr, xdbl,
            ML, XPROJ_N, DINNER, DINNER, DINNER, XPROJ_N);

        // 5b) repack dt input: xdbl[:, :12] -> dtp16[ML,32] zero-padded
        pad_cvt_kernel<<<(ML * DTK + ew - 1) / ew, ew, 0, stream>>>(
            xdbl, dtp16, ML, XPROJ_N, DTRANK, DTK);

        // 6) dt_proj: dtf[ML,384] = dtp16 @ dtw16^T   (WMMA, K=32, no tail)
        wmma_gemm<<<gemm_grid(ML, DINNER), 256, 0, stream>>>(
            dtp16, dtw16 + (long)i * DINNER * DTK, nullptr, dtf,
            ML, DINNER, DTK, DTK, DTK, DINNER);

        // 7) selective scan -> y16
        mamba_scan_kernel<<<(BATCH * DINNER + ew - 1) / ew, ew, 0, stream>>>(
            dtf, dt_b + (long)i * DINNER, xz, u2, xdbl,
            A_log + (long)i * DINNER * DSTATE, D_param + (long)i * DINNER, y16);

        // 8) out_proj: hid[ML,192] = y16 @ out_w^T   (WMMA, K=384)
        wmma_gemm<<<gemm_grid(ML, DMODEL), 256, 0, stream>>>(
            y16, outw16 + (long)i * DMODEL * DINNER, nullptr, hid,
            ML, DMODEL, DINNER, DINNER, DINNER, DMODEL);
    }

    // 9) final RMSNorm(hidden + residual) -> hbuf16
    add_rmsnorm_kernel<<<ML, 256, 0, stream>>>(hid, resid, resid, hbuf16, normf_w);

    // 10) gather cls rows -> cls16 [32,192]
    gather_cls_kernel<<<(BATCH * DMODEL + ew - 1) / ew, ew, 0, stream>>>(hbuf16, cls16);

    // 11) head: out[32,1000] = cls16 @ head_w^T + head_b   (WMMA, K=192)
    wmma_gemm<<<gemm_grid(BATCH, 1000), 256, 0, stream>>>(
        cls16, hdw16, head_b, out,
        BATCH, 1000, DMODEL, DMODEL, DMODEL, 1000);

    (void)in_sizes; (void)n_in; (void)out_size; (void)ws_size;
}
